// TopKGate_89043261980986
// MI455X (gfx1250) — matrix operand order repaired and checked by reading, not compiled, not run on hardware
//
#include <hip/hip_runtime.h>
#include <hip/hip_bf16.h>

// MoE threshold-gating for MI455X (gfx1250, wave32, WMMA, async-LDS DMA).
//
// Roofline: GEMM reads x = 134 MB -> ~5.8 us HBM floor @ 23.3 TB/s; FLOPs are
// only 4.3 G. bf16x3 split-precision on v_wmma_f32_16x16x32_bf16 keeps the
// matrix pipe under the memory floor at ~fp32 accuracy. Global->LDS staging
// uses GLOBAL_LOAD_ASYNC_TO_LDS_B128 (ASYNCcnt) so the HBM stream of chunk
// k+1 overlaps the WMMAs of chunk k inside each workgroup. Logits never
// round-trip to HBM: gating is fused into the GEMM epilogue in LDS.

#define NTOK   16384
#define KDIM   2048
#define NEXP   64
#define CAP    256        // ceil(NTOK/NEXP * capacity_factor)
#define GTHR   0.5f       // gating threshold

#define KC     128                 // K staging chunk
#define ROWP   (KC + 8)            // padded LDS row (ushorts): 272B, 16B-aligned, bank-skewed

typedef __bf16 v16bf __attribute__((ext_vector_type(16)));
typedef float  v8f   __attribute__((ext_vector_type(8)));
typedef unsigned int u32x4 __attribute__((ext_vector_type(4)));

union Frag { u32x4 q[2]; v16bf v; };

static __device__ inline v8f wmma_bf16(v16bf a, v16bf b, v8f c) {
  // (neg_a, A, neg_b, B, c_mod, C, reuse_a, reuse_b) -> v_wmma_f32_16x16x32_bf16
  return __builtin_amdgcn_wmma_f32_16x16x32_bf16(false, a, false, b, (short)0, c, false, false);
}

// Assemble a 16-element bf16 fragment from two 16-byte LDS runs.
static __device__ inline v16bf ld_frag(const unsigned short* p0, const unsigned short* p1) {
  Frag f;
  f.q[0] = *(const u32x4*)p0;
  f.q[1] = *(const u32x4*)p1;
  return f.v;
}

// fp32 -> bf16 hi/lo split (RNE both rounds): a ~= hi + lo, ~16 mantissa bits kept.
static __device__ inline void cvt_split(float f, unsigned short* hi, unsigned short* lo) {
  __bf16 h = (__bf16)f;
  float hf = (float)h;
  __bf16 l = (__bf16)(f - hf);
  *hi = __builtin_bit_cast(unsigned short, h);
  *lo = __builtin_bit_cast(unsigned short, l);
}

// CDNA5 async global->LDS DMA, 16B per lane, GVS addressing:
//   each lane's 16 B at SADDR + VADDR go to LDS at VDST   (tracked by ASYNCcnt)
// lds_off: LDS byte address (low 32 bits of the flat shared pointer, per the
// ISA's flat->LDS aperture rule).
static __device__ inline void async_ld16(unsigned lds_off, int gvoff, const void* sbase) {
  asm volatile("global_load_async_to_lds_b128 %0, %1, %2"
               :: "v"(lds_off), "v"(gvoff), "s"(sbase)
               : "memory");
}
static __device__ inline void wait_async0() {
  asm volatile("s_wait_asynccnt 0x0" ::: "memory");
}

// ---------------------------------------------------------------------------
// Kernel 1: fused GEMM (bf16x3 WMMA) + softmax + threshold ranking.
// Grid: NTOK/16 blocks of 128 threads (4 waves). Wave w computes the 16x16
// tile for experts [16w,16w+16) of this block's 16 tokens.
// ---------------------------------------------------------------------------
__global__ __launch_bounds__(128) void moe_gate_kernel(
    const float* __restrict__ x, const float* __restrict__ Wg,
    float* __restrict__ tamT, float* __restrict__ gatesT,
    float* __restrict__ sumg, float* __restrict__ cntg)
{
  // fp32 async stage: rows 0..15 = x tile, rows 16..79 = W tile  (40960 B)
  __shared__ alignas(16) float stage[80 * KC];
  // bf16 hi/lo tiles                                            (43520 B)
  __shared__ alignas(16) unsigned short bfbuf[160 * ROWP];
  unsigned short* xs_hi = bfbuf;                 // [16][ROWP]
  unsigned short* xs_lo = xs_hi + 16 * ROWP;     // [16][ROWP]
  unsigned short* w_hi  = xs_lo + 16 * ROWP;     // [64][ROWP]
  unsigned short* w_lo  = w_hi  + 64 * ROWP;     // [64][ROWP]

  const int tid   = threadIdx.x;
  const int wv    = tid >> 5;          // wave 0..3
  const int lane  = tid & 31;
  const int row   = lane & 15;         // A: M index / B: N index
  const int half  = lane >> 4;
  const int e0    = wv * 16;           // expert tile base
  const int tok0  = blockIdx.x * 16;
  const int abase = half * 16;         // A K-base per lane half (contiguous K, matches B)
  const int bbase = half * 16;         // B K-base per lane half (contiguous K, matches A)

  v8f c = {0.f, 0.f, 0.f, 0.f, 0.f, 0.f, 0.f, 0.f};

  // Issue async DMA of one K-chunk into the fp32 stage buffer.
  // Trip counts are multiples of 128 -> uniform control flow for all lanes.
  auto issue_chunk = [&](int kc) {
    for (int i = tid; i < (16 * KC) / 4; i += 128) {          // 4 per thread
      int r  = i / (KC / 4);
      int c4 = (i % (KC / 4)) * 4;
      unsigned lds = (unsigned)(uintptr_t)(const void*)(stage + r * KC + c4);
      int gv = ((tok0 + r) * KDIM + kc + c4) * 4;             // < 2^31
      async_ld16(lds, gv, x);
    }
    for (int i = tid; i < (64 * KC) / 4; i += 128) {          // 16 per thread
      int r  = i / (KC / 4);
      int c4 = (i % (KC / 4)) * 4;
      unsigned lds = (unsigned)(uintptr_t)(const void*)(stage + (16 + r) * KC + c4);
      int gv = (r * KDIM + kc + c4) * 4;
      async_ld16(lds, gv, Wg);
    }
  };

  issue_chunk(0);   // prologue

  for (int kc = 0; kc < KDIM; kc += KC) {
    wait_async0();        // own-wave ASYNCcnt==0; same i-mapping as convert below
    __syncthreads();      // all waves' DMA complete -> stage fully valid

    // ---- convert stage fp32 -> bf16 hi/lo tiles (LDS->VALU->LDS) ----
    for (int i = tid; i < (16 * KC) / 4; i += 128) {
      int r  = i / (KC / 4);
      int c4 = (i % (KC / 4)) * 4;
      float4 v = *(const float4*)(stage + r * KC + c4);
      unsigned short* ph = xs_hi + r * ROWP + c4;
      unsigned short* pl = xs_lo + r * ROWP + c4;
      cvt_split(v.x, ph + 0, pl + 0);
      cvt_split(v.y, ph + 1, pl + 1);
      cvt_split(v.z, ph + 2, pl + 2);
      cvt_split(v.w, ph + 3, pl + 3);
    }
    for (int i = tid; i < (64 * KC) / 4; i += 128) {
      int r  = i / (KC / 4);
      int c4 = (i % (KC / 4)) * 4;
      float4 v = *(const float4*)(stage + (16 + r) * KC + c4);
      unsigned short* ph = w_hi + r * ROWP + c4;
      unsigned short* pl = w_lo + r * ROWP + c4;
      cvt_split(v.x, ph + 0, pl + 0);
      cvt_split(v.y, ph + 1, pl + 1);
      cvt_split(v.z, ph + 2, pl + 2);
      cvt_split(v.w, ph + 3, pl + 3);
    }
    __syncthreads();      // bf16 tiles ready; stage fully drained

    // ---- overlap: DMA of chunk k+1 runs under the WMMAs of chunk k ----
    if (kc + KC < KDIM) issue_chunk(kc + KC);

    // ---- K loop: 32 per step, bf16x3 -> 3 WMMAs ----
    const unsigned short* xah = xs_hi + row * ROWP;
    const unsigned short* xal = xs_lo + row * ROWP;
    const unsigned short* wbh = w_hi + (e0 + row) * ROWP;
    const unsigned short* wbl = w_lo + (e0 + row) * ROWP;
#pragma unroll
    for (int kk = 0; kk < KC; kk += 32) {
      // A 16x32 bf16: lane holds M=row, K=[abase..abase+15] contiguous
      v16bf ah = ld_frag(xah + kk + abase, xah + kk + abase + 8);
      v16bf al = ld_frag(xal + kk + abase, xal + kk + abase + 8);
      // B 32x16 bf16: lane holds N=row, K=[bbase..bbase+15] contiguous
      v16bf bh = ld_frag(wbh + kk + bbase, wbh + kk + bbase + 8);
      v16bf bl = ld_frag(wbl + kk + bbase, wbl + kk + bbase + 8);
      c = wmma_bf16(ah, bl, c);   // small terms first
      c = wmma_bf16(al, bh, c);
      c = wmma_bf16(ah, bh, c);
    }
  }
  __syncthreads();

  // ---- epilogue: logits tile to LDS (reuse stage buffer) ----
  float* lg     = stage;               // [16][68] padded rows -> conflict-free
  float* ce_loc = lg + 16 * 68;        // [64] per-WG argmax counts
#pragma unroll
  for (int j = 0; j < 8; ++j)          // C layout: VGPR j -> M = j + half*8, N = row
    lg[(half * 8 + j) * 68 + e0 + row] = c[j];
  if (tid < 64) ce_loc[tid] = 0.f;
  __syncthreads();

  // ---- gating: 8 threads per token, each owns experts {sub, sub+8, ...} ----
  const int token = tid >> 3;
  const int sub   = tid & 7;
  float* g = lg + token * 68;

  float ge[8];
  float m = -3.402823466e38f; int mi = 0;
#pragma unroll
  for (int i = 0; i < 8; ++i) {
    int e = sub + 8 * i;
    ge[i] = g[e];
    if (ge[i] > m) { m = ge[i]; mi = e; }   // strictly-greater keeps lowest index
  }
#pragma unroll
  for (int off = 4; off > 0; off >>= 1) {   // 8-lane groups within wave32
    float om = __shfl_xor(m, off, 8);
    int   oi = __shfl_xor(mi, off, 8);
    if (om > m || (om == m && oi < mi)) { m = om; mi = oi; }
  }
  if (sub == 0) atomicAdd(&ce_loc[mi], 1.0f);   // ds_add_f32

  float ex[8]; float s = 0.f;
#pragma unroll
  for (int i = 0; i < 8; ++i) { ex[i] = expf(ge[i] - m); s += ex[i]; }
#pragma unroll
  for (int off = 4; off > 0; off >>= 1) s += __shfl_xor(s, off, 8);
  const float inv = 1.0f / s;
  __syncthreads();
#pragma unroll
  for (int i = 0; i < 8; ++i) { ge[i] = ex[i] * inv; g[sub + 8 * i] = ge[i]; }
  __syncthreads();

  // per-expert partial sums for l_aux (one global atomic per expert per WG)
  if (tid < 64) {
    float acc = 0.f;
#pragma unroll
    for (int t = 0; t < 16; ++t) acc += lg[t * 68 + tid];
    atomicAdd(&sumg[tid], acc);
    float cl = ce_loc[tid];
    if (cl != 0.f) atomicAdd(&cntg[tid], cl);
  }

  // rank/prefix == sorted-cumsum threshold test (stable sort, index tie-break):
  // rank = #{j sorted before e}; prefix = sum of their gates; chosen iff
  // rank==0 (forced top-1) or prefix < 0.5. Importance = (E - rank) + gate.
  int   rk[8] = {0,0,0,0,0,0,0,0};
  float pf[8] = {0,0,0,0,0,0,0,0};
  for (int j = 0; j < 64; ++j) {
    float gj = g[j];
#pragma unroll
    for (int i = 0; i < 8; ++i) {
      int e = sub + 8 * i;
      bool before = (gj > ge[i]) || (gj == ge[i] && j < e);
      rk[i] += before ? 1 : 0;
      pf[i] += before ? gj : 0.f;
    }
  }
  const int n = tok0 + token;
#pragma unroll
  for (int i = 0; i < 8; ++i) {
    int e = sub + 8 * i;
    bool chosen = (rk[i] == 0) || (pf[i] < GTHR);
    float tam = chosen ? (float)(NEXP - rk[i]) + ge[i] : 0.f;
    tamT[(size_t)e * NTOK + n]   = tam;     // transposed -> coalesced column scans
    gatesT[(size_t)e * NTOK + n] = ge[i];
  }
}

// ---------------------------------------------------------------------------
// Kernel 2: per-expert capacity threshold = 256th-largest tam in the column.
// Two-level LDS histogram on importance in [1,65): 1/16 bins, then 1/16384.
// ---------------------------------------------------------------------------
__global__ __launch_bounds__(256) void expert_thresh_kernel(
    const float* __restrict__ tamT, float* __restrict__ thr)
{
  const int e = blockIdx.x;
  const float* col = tamT + (size_t)e * NTOK;
  __shared__ unsigned int hist[1056];
  __shared__ int sB; __shared__ int sNeed;

  for (int i = threadIdx.x; i < 1056; i += 256) hist[i] = 0u;
  __syncthreads();
  for (int n = threadIdx.x; n < NTOK; n += 256) {
    float v = col[n];
    if (v > 0.f) {
      int b = (int)(v * 16.f); if (b > 1055) b = 1055;
      atomicAdd(&hist[b], 1u);
    }
  }
  __syncthreads();
  if (threadIdx.x == 0) {
    unsigned int acc = 0; int B = -1; unsigned int cgt = 0;
    for (int b = 1055; b >= 0; --b) {
      unsigned int nb = acc + hist[b];
      if (nb >= (unsigned)CAP) { B = b; cgt = acc; break; }
      acc = nb;
    }
    sB = B; sNeed = CAP - (int)cgt;
  }
  __syncthreads();
  const int B = sB;
  if (B < 0) {                      // column not over capacity: keep all chosen
    if (threadIdx.x == 0) thr[e] = 0.5f;   // chosen tam >= 1 always passes
    return;
  }
  for (int i = threadIdx.x; i < 1024; i += 256) hist[i] = 0u;
  __syncthreads();
  for (int n = threadIdx.x; n < NTOK; n += 256) {
    float v = col[n];
    if (v > 0.f) {
      int b = (int)(v * 16.f); if (b > 1055) b = 1055;
      if (b == B) {
        int sbin = (int)((v * 16.f - (float)B) * 1024.f);
        sbin = sbin < 0 ? 0 : (sbin > 1023 ? 1023 : sbin);
        atomicAdd(&hist[sbin], 1u);
      }
    }
  }
  __syncthreads();
  if (threadIdx.x == 0) {
    const int need = sNeed;
    unsigned int acc = 0; int S = 0;
    for (int s2 = 1023; s2 >= 0; --s2) {
      acc += hist[s2];
      if ((int)acc >= need) { S = s2; break; }
    }
    thr[e] = ((float)B + (float)S * (1.f / 1024.f)) * (1.f / 16.f);
  }
}

// ---------------------------------------------------------------------------
// Kernel 3: combine_weights = gates * (tam >= thr[e]).  out row-major [N,E].
// ---------------------------------------------------------------------------
__global__ __launch_bounds__(256) void combine_kernel(
    const float* __restrict__ gatesT, const float* __restrict__ tamT,
    const float* __restrict__ thr, float* __restrict__ out)
{
  const int i = blockIdx.x * 256 + threadIdx.x;   // i = e*NTOK + n (coalesced reads)
  const int e = i >> 14;
  const int n = i & (NTOK - 1);
  const float keep = (tamT[i] >= thr[e]) ? 1.0f : 0.0f;
  out[1 + (size_t)n * NEXP + e] = gatesT[i] * keep;
}

// Kernel 4: l_aux = E * sum_e mean(gates)_e * mean(onehot_argmax)_e
__global__ __launch_bounds__(64) void laux_kernel(
    const float* __restrict__ sumg, const float* __restrict__ cntg,
    float* __restrict__ out)
{
  __shared__ float r[64];
  const int t = threadIdx.x;
  r[t] = sumg[t] * cntg[t];
  __syncthreads();
  if (t == 0) {
    float s = 0.f;
    for (int i = 0; i < 64; ++i) s += r[i];
    out[0] = s * (float)NEXP / ((float)NTOK * (float)NTOK);
  }
}

__global__ void init_kernel(float* p) { p[threadIdx.x] = 0.f; }  // 128 floats

extern "C" void kernel_launch(void* const* d_in, const int* in_sizes, int n_in,
                              void* d_out, int out_size, void* d_ws, size_t ws_size,
                              hipStream_t stream) {
  (void)in_sizes; (void)n_in; (void)out_size; (void)ws_size;
  const float* x  = (const float*)d_in[0];   // [16384, 2048] fp32
  const float* Wg = (const float*)d_in[1];   // [64, 2048] fp32
  float* out = (float*)d_out;                // [1 + 16384*64] fp32

  // Workspace layout (~8.4 MB): tamT | gatesT | sumg | cntg | thr
  float* tamT   = (float*)d_ws;                        // 64*16384
  float* gatesT = tamT + (size_t)NEXP * NTOK;          // 64*16384
  float* sumg   = gatesT + (size_t)NEXP * NTOK;        // 64
  float* cntg   = sumg + 64;                           // 64
  float* thr    = cntg + 64;                           // 64

  init_kernel<<<1, 128, 0, stream>>>(sumg);            // zero sumg+cntg (contiguous)
  moe_gate_kernel<<<NTOK / 16, 128, 0, stream>>>(x, Wg, tamT, gatesT, sumg, cntg);
  expert_thresh_kernel<<<NEXP, 256, 0, stream>>>(tamT, thr);
  laux_kernel<<<1, 64, 0, stream>>>(sumg, cntg, out);
  combine_kernel<<<(NTOK * NEXP) / 256, 256, 0, stream>>>(gatesT, tamT, thr, out);
}
